// DecoderLayer_62491774157505
// MI455X (gfx1250) — matrix operand order repaired and checked by reading, not compiled, not run on hardware
//
#include <hip/hip_runtime.h>
#include <math.h>

typedef __bf16 bf16_t;
typedef __attribute__((ext_vector_type(16))) __bf16 v16bf;
typedef __attribute__((ext_vector_type(8)))  __bf16 v8bf;
typedef __attribute__((ext_vector_type(8)))  float  v8f;

union FragBF { v16bf v; v8bf h[2]; };

__device__ __forceinline__ bf16_t f2bf(float f) {
  union { float f; unsigned u; } a; a.f = f;
  unsigned r = a.u + 0x7FFFu + ((a.u >> 16) & 1u);
  union { unsigned short s; bf16_t b; } o; o.s = (unsigned short)(r >> 16);
  return o.b;
}
__device__ __forceinline__ v8bf zero_v8bf() {
  union { unsigned u[4]; v8bf v; } z; z.u[0] = z.u[1] = z.u[2] = z.u[3] = 0u;
  return z.v;
}

// ---------------------------------------------------------------------------
// Generic batched WMMA GEMM: C[z] = A[z] (MxK, bf16) * B[z] (KxN, bf16) + bias
// Batch offset per operand: (z / nb2) * s1 + (z % nb2) * s2   (element units)
// EPI: 0 = f32 out, 1 = bf16 out, 2 = exact-GELU -> bf16, 3 = f32 out + resid
// BN: 128 (big GEMMs, 8 WMMA/wave/step) or 64 (attention GEMMs)
// Requires: K % 32 == 0, N % BN == 0.
// ---------------------------------------------------------------------------
struct GemmArgs {
  const bf16_t* A; const bf16_t* B; void* C; const float* bias; const float* R;
  long lda, ldb, ldc, ldr;
  long sA1, sA2, sB1, sB2, sC1, sC2, sR1, sR2;
  int nbA2, nbB2, nbC2, nbR2;
  int M, N, K;
};

template<int EPI, int BN>
__global__ void __launch_bounds__(256) gemm_wmma_k(GemmArgs g)
{
  constexpr int NJ = BN / 32;          // 16-col frag tiles per wave
  const int z = blockIdx.z;
  const bf16_t* A  = g.A + (long)(z / g.nbA2) * g.sA1 + (long)(z % g.nbA2) * g.sA2;
  const bf16_t* Bm = g.B + (long)(z / g.nbB2) * g.sB1 + (long)(z % g.nbB2) * g.sB2;
  const long zC = (long)(z / g.nbC2) * g.sC1 + (long)(z % g.nbC2) * g.sC2;
  float*  Cf = (float*)g.C  + zC;
  bf16_t* Cb = (bf16_t*)g.C + zC;
  const float* Rp = g.R ? (g.R + (long)(z / g.nbR2) * g.sR1 + (long)(z % g.nbR2) * g.sR2)
                        : (const float*)nullptr;

  const int m0 = blockIdx.y * 128;
  const int n0 = blockIdx.x * BN;
  const int tid  = threadIdx.x;
  const int lane = tid & 31;
  const int wv   = tid >> 5;
  const int wm   = (wv & 3) * 32;           // wave row offset in block tile
  const int wn   = (wv >> 2) * (BN / 2);    // wave col offset in block tile
  const bool fullM = (m0 + 128 <= g.M);     // uniform: async fast path legal

  __shared__ __attribute__((aligned(16))) bf16_t sA[2][128 * 32]; // [m][k] row-major
  __shared__ __attribute__((aligned(16))) bf16_t sB[2][BN * 32];  // [n][k] (K-transposed)

  v8f acc[2][NJ];
  {
    v8f zz = {0.f, 0.f, 0.f, 0.f, 0.f, 0.f, 0.f, 0.f};
#pragma unroll
    for (int i = 0; i < 2; i++)
#pragma unroll
      for (int j = 0; j < NJ; j++) acc[i][j] = zz;
  }

  const int ktiles = g.K / 32;

  auto load_tile = [&](int kt, int buf) {
    const long k0 = (long)kt * 32;
    // ---- A tile: 128 rows x 32 cols ----
    if (fullM) {
      // gfx1250 async copy straight into LDS (ASYNCcnt-tracked, no VGPR bounce)
#pragma unroll
      for (int i = 0; i < 2; i++) {
        int c  = tid + i * 256;
        int r  = c >> 2;
        int cc = (c & 3) * 8;
        unsigned lds_addr = (unsigned)(size_t)&sA[buf][r * 32 + cc];
        unsigned long long ga =
            (unsigned long long)(const void*)(A + (long)(m0 + r) * g.lda + k0 + cc);
        asm volatile("global_load_async_to_lds_b128 %0, %1, off"
                     :: "v"(lds_addr), "v"(ga) : "memory");
      }
    } else {
#pragma unroll
      for (int i = 0; i < 2; i++) {
        int c  = tid + i * 256;
        int r  = c >> 2;
        int cc = (c & 3) * 8;
        v8bf val = zero_v8bf();
        if (m0 + r < g.M)
          val = *(const v8bf*)(A + (long)(m0 + r) * g.lda + k0 + cc);
        *(v8bf*)&sA[buf][r * 32 + cc] = val;
      }
    }
    // ---- B tile: 32 rows x BN cols, coalesced reads, transposed LDS writes ----
#pragma unroll
    for (int i = 0; i < BN / 64; i++) {
      int c  = tid + i * 256;
      int kr = c / (BN / 8);
      int cc = (c % (BN / 8)) * 8;
      v8bf val = *(const v8bf*)(Bm + (k0 + kr) * g.ldb + n0 + cc);
#pragma unroll
      for (int e = 0; e < 8; e++)
        sB[buf][(cc + e) * 32 + kr] = val[e];
    }
  };

  load_tile(0, 0);
  asm volatile("s_wait_asynccnt 0x0" ::: "memory");
  __syncthreads();

  for (int kt = 0; kt < ktiles; ++kt) {
    const int buf = kt & 1;
    if (kt + 2 < ktiles) {  // gfx1250 global_prefetch_b8 of tile kt+2
      __builtin_prefetch(Bm + (long)(kt + 2) * 32 * g.ldb + n0, 0, 1);
      __builtin_prefetch(A + (long)m0 * g.lda + (long)(kt + 2) * 32, 0, 1);
    }
    if (kt + 1 < ktiles) load_tile(kt + 1, buf ^ 1);

    // Fragment loads per ISA VGPR layouts (wave32)
    {
      const int lr  = lane & 15;
      const int kbA = (lane < 16) ? 0 : 8;   // A: K 0-7/16-23 vs 8-15/24-31
      const int kbB = (lane < 16) ? 0 : 8;   // B: K 0-7/16-23 vs 8-15/24-31
      FragBF a[2], b[NJ];
#pragma unroll
      for (int i = 0; i < 2; i++) {
        int row = wm + i * 16 + lr;
        a[i].h[0] = *(const v8bf*)&sA[buf][row * 32 + kbA];
        a[i].h[1] = *(const v8bf*)&sA[buf][row * 32 + kbA + 16];
      }
#pragma unroll
      for (int j = 0; j < NJ; j++) {
        int col = wn + j * 16 + lr;
        b[j].h[0] = *(const v8bf*)&sB[buf][col * 32 + kbB];
        b[j].h[1] = *(const v8bf*)&sB[buf][col * 32 + kbB + 16];
      }
#pragma unroll
      for (int i = 0; i < 2; i++)
#pragma unroll
        for (int j = 0; j < NJ; j++)
          acc[i][j] = __builtin_amdgcn_wmma_f32_16x16x32_bf16(
              false, a[i].v, false, b[j].v, (short)0, acc[i][j], false, false);
    }
    // complete next tile's async A copy before the publish barrier
    asm volatile("s_wait_asynccnt 0x0" ::: "memory");
    __syncthreads();
  }

  // Epilogue: C VGPR layout -> m = tileRow + (lane/16)*8 + e, n = tileCol + lane%16
  const int lr = lane & 15;
  const int lh = lane >> 4;
#pragma unroll
  for (int i = 0; i < 2; i++) {
#pragma unroll
    for (int j = 0; j < NJ; j++) {
      const int nn = n0 + wn + j * 16 + lr;
      const float bv = g.bias ? g.bias[nn] : 0.0f;
      const int mb = m0 + wm + i * 16 + lh * 8;
#pragma unroll
      for (int e = 0; e < 8; e++) {
        const int m = mb + e;
        if (m < g.M) {
          float v = acc[i][j][e] + bv;
          const long co = (long)m * g.ldc + nn;
          if (EPI == 0) {
            Cf[co] = v;
          } else if (EPI == 1) {
            Cb[co] = f2bf(v);
          } else if (EPI == 2) {
            float ge = 0.5f * v * (1.0f + erff(v * 0.70710678118654752f));
            Cb[co] = f2bf(ge);
          } else {
            Cf[co] = v + Rp[(long)m * g.ldr + nn];
          }
        }
      }
    }
  }
}

// ---------------------------------------------------------------------------
// LayerNorm (+ optional FiLM + SiLU) -> bf16.  mode 0: LN only; mode 1: FiLM+SiLU
// ---------------------------------------------------------------------------
__global__ void ln_film_k(const float* __restrict__ x, const float* __restrict__ g,
                          const float* __restrict__ bta, const float* __restrict__ eo,
                          int rowsPerBatch, bf16_t* __restrict__ out, int cols, int mode)
{
  const long row = blockIdx.x;
  const float* xr = x + row * cols;
  float s = 0.f, sq = 0.f;
  for (int c = threadIdx.x; c < cols; c += 256) { float v = xr[c]; s += v; sq += v * v; }
#pragma unroll
  for (int o = 16; o > 0; o >>= 1) { s += __shfl_xor(s, o, 32); sq += __shfl_xor(sq, o, 32); }
  __shared__ float sh_s[8], sh_q[8];
  const int lane = threadIdx.x & 31, w = threadIdx.x >> 5;
  if (lane == 0) { sh_s[w] = s; sh_q[w] = sq; }
  __syncthreads();
  float ts = 0.f, tq = 0.f;
#pragma unroll
  for (int i = 0; i < 8; i++) { ts += sh_s[i]; tq += sh_q[i]; }
  const float mean = ts / cols;
  const float var  = tq / cols - mean * mean;
  const float rstd = rsqrtf(var + 1e-5f);
  const int b = (int)(row / rowsPerBatch);
  const float* eob = eo ? (eo + (long)b * 2 * cols) : (const float*)nullptr;
  bf16_t* orow = out + row * cols;
  for (int c = threadIdx.x; c < cols; c += 256) {
    float h = (xr[c] - mean) * rstd * g[c] + bta[c];
    if (mode == 1) {
      h = h * (1.0f + eob[c]) + eob[cols + c];
      h = h * (1.0f / (1.0f + expf(-h)));      // SiLU
    }
    orow[c] = f2bf(h);
  }
}

// softmax over last dim (hd = 64), one wave per row -> bf16
__global__ void softmax_hd_k(const float* __restrict__ in, bf16_t* __restrict__ out, long rows)
{
  const long row = (long)blockIdx.x * 8 + (threadIdx.x >> 5);
  if (row >= rows) return;
  const int lane = threadIdx.x & 31;
  const float* p = in + row * 64;
  float a = p[lane], b = p[lane + 32];
  float mx = fmaxf(a, b);
#pragma unroll
  for (int o = 16; o > 0; o >>= 1) mx = fmaxf(mx, __shfl_xor(mx, o, 32));
  float ea = expf(a - mx), eb = expf(b - mx);
  float s = ea + eb;
#pragma unroll
  for (int o = 16; o > 0; o >>= 1) s += __shfl_xor(s, o, 32);
  const float inv = 1.0f / s;
  bf16_t* q = out + row * 64;
  q[lane] = f2bf(ea * inv);
  q[lane + 32] = f2bf(eb * inv);
}

// softmax over token axis with optional mask, writing k^T [B,H,hd,Tn] (bf16).
// Layout of kin is [B,Tn,H=16,hd=64] with row stride 1024. 64 threads/block.
__global__ void softmax_t_tr_k(const float* __restrict__ kin, const float* __restrict__ mask,
                               bf16_t* __restrict__ kT, int Tn)
{
  const int bh = blockIdx.x;
  const int b = bh >> 4, h = bh & 15;
  const int d = threadIdx.x;
  const float* base = kin + (long)b * Tn * 1024 + h * 64 + d;
  const float* mrow = mask ? (mask + (long)b * Tn) : (const float*)nullptr;
  float mx = -3.0e38f;
  for (int t = 0; t < Tn; ++t) {
    float v = base[(long)t * 1024];
    if (mrow) v += (1.0f - mrow[t]) * -1.0e6f;
    mx = fmaxf(mx, v);
  }
  float s = 0.f;
  for (int t = 0; t < Tn; ++t) {
    float v = base[(long)t * 1024];
    if (mrow) v += (1.0f - mrow[t]) * -1.0e6f;
    s += expf(v - mx);
  }
  const float inv = 1.0f / s;
  bf16_t* orow = kT + ((long)bh * 64 + d) * Tn;
  for (int t = 0; t < Tn; ++t) {
    float v = base[(long)t * 1024];
    if (mrow) v += (1.0f - mrow[t]) * -1.0e6f;
    orow[t] = f2bf(expf(v - mx) * inv);
  }
}

// elementwise helpers (grid-stride)
__global__ void cvt_bf16_k(const float* __restrict__ in, bf16_t* __restrict__ out, long n)
{
  long i = (long)blockIdx.x * blockDim.x + threadIdx.x;
  const long st = (long)gridDim.x * blockDim.x;
  for (; i < n; i += st) out[i] = f2bf(in[i]);
}
__global__ void maskmul_bf16_k(const float* __restrict__ in, const float* __restrict__ mask,
                               bf16_t* __restrict__ out, long n, int cols)
{
  long i = (long)blockIdx.x * blockDim.x + threadIdx.x;
  const long st = (long)gridDim.x * blockDim.x;
  for (; i < n; i += st) {
    float m = mask ? mask[i / cols] : 1.0f;
    out[i] = f2bf(in[i] * m);
  }
}
__global__ void silu_bf16_k(const float* __restrict__ in, bf16_t* __restrict__ out, long n)
{
  long i = (long)blockIdx.x * blockDim.x + threadIdx.x;
  const long st = (long)gridDim.x * blockDim.x;
  for (; i < n; i += st) {
    float v = in[i];
    out[i] = f2bf(v * (1.0f / (1.0f + expf(-v))));
  }
}

// ---------------------------------------------------------------------------
// Host side
// ---------------------------------------------------------------------------
template<int BN>
static void run_gemm_bn(hipStream_t st, int epi, const GemmArgs& g, int batch)
{
  dim3 grid(g.N / BN, (g.M + 127) / 128, batch), blk(256);
  switch (epi) {
    case 0: gemm_wmma_k<0, BN><<<grid, blk, 0, st>>>(g); break;
    case 1: gemm_wmma_k<1, BN><<<grid, blk, 0, st>>>(g); break;
    case 2: gemm_wmma_k<2, BN><<<grid, blk, 0, st>>>(g); break;
    default: gemm_wmma_k<3, BN><<<grid, blk, 0, st>>>(g); break;
  }
}
static void run_gemm(hipStream_t st, int epi, GemmArgs g, int batch, int bn)
{
  if (g.nbA2 < 1) g.nbA2 = 1;
  if (g.nbB2 < 1) g.nbB2 = 1;
  if (g.nbC2 < 1) g.nbC2 = 1;
  if (g.nbR2 < 1) g.nbR2 = 1;
  if (bn == 128) run_gemm_bn<128>(st, epi, g, batch);
  else           run_gemm_bn<64>(st, epi, g, batch);
}

static GemmArgs mk2d(const bf16_t* A, long lda, const bf16_t* B, long ldb,
                     void* C, long ldc, const float* bias,
                     const float* R, long ldr, int M, int N, int K)
{
  GemmArgs g;
  g.A = A; g.B = B; g.C = C; g.bias = bias; g.R = R;
  g.lda = lda; g.ldb = ldb; g.ldc = ldc; g.ldr = ldr;
  g.sA1 = g.sA2 = g.sB1 = g.sB2 = g.sC1 = g.sC2 = g.sR1 = g.sR2 = 0;
  g.nbA2 = g.nbB2 = g.nbC2 = g.nbR2 = 1;
  g.M = M; g.N = N; g.K = K;
  return g;
}

extern "C" void kernel_launch(void* const* d_in, const int* in_sizes, int n_in,
                              void* d_out, int out_size, void* d_ws, size_t ws_size,
                              hipStream_t stream)
{
  (void)in_sizes; (void)n_in; (void)out_size; (void)ws_size;
  constexpr int  Bc = 16, Tc = 1024, Nc = 256, Dc = 1024, Dtc = 768, TEc = 2048,
                 FFc = 4096, Hc = 16;
  constexpr long BT = (long)Bc * Tc;      // 16384
  constexpr long BNr = (long)Bc * Nc;     // 4096

  const float* x    = (const float*)d_in[0];
  const float* xf   = (const float*)d_in[1];
  const float* emb  = (const float*)d_in[2];
  const float* mask = (const float*)d_in[3];
  const float* P[40];
  for (int i = 0; i < 40; i++) P[i] = (const float*)d_in[4 + i];
  // P indices: 0..7 sa ln/qkv, 8..13 sa_st, 14..17 ca ln/tln, 18..23 ca qkv,
  //            24..29 ca_st, 30..33 ffn, 34..39 ff_st

  // ---- workspace carving ----
  char* wsb = (char*)d_ws;
  size_t off = 0;
  auto carve = [&](size_t bytes) -> char* {
    char* r = wsb + off;
    off += (bytes + 255) & ~(size_t)255;
    return r;
  };
  struct WRec { int idx; long n; bf16_t* dst; };
  WRec wr[14] = {
    { 2, (long)Dc * Dc, nullptr},      // sa_q_w
    { 4, (long)Dc * Dc, nullptr},      // sa_k_w
    { 6, (long)Dc * Dc, nullptr},      // sa_v_w
    { 8, (long)TEc * 2 * Dc, nullptr}, // sa_st_emb_w
    {12, (long)Dc * Dc, nullptr},      // sa_st_out_w
    {18, (long)Dc * Dc, nullptr},      // ca_q_w
    {20, (long)Dtc * Dc, nullptr},     // ca_k_w
    {22, (long)Dtc * Dc, nullptr},     // ca_v_w
    {24, (long)TEc * 2 * Dc, nullptr}, // ca_st_emb_w
    {28, (long)Dc * Dc, nullptr},      // ca_st_out_w
    {30, (long)Dc * FFc, nullptr},     // ff1_w
    {32, (long)FFc * Dc, nullptr},     // ff2_w
    {34, (long)TEc * 2 * Dc, nullptr}, // ff_st_emb_w
    {38, (long)Dc * Dc, nullptr},      // ff_st_out_w
  };
  for (int i = 0; i < 14; i++) wr[i].dst = (bf16_t*)carve(wr[i].n * 2);
  bf16_t* w_saq = wr[0].dst;  bf16_t* w_sak = wr[1].dst;  bf16_t* w_sav = wr[2].dst;
  bf16_t* w_saemb = wr[3].dst; bf16_t* w_saout = wr[4].dst;
  bf16_t* w_caq = wr[5].dst;  bf16_t* w_cak = wr[6].dst;  bf16_t* w_cav = wr[7].dst;
  bf16_t* w_caemb = wr[8].dst; bf16_t* w_caout = wr[9].dst;
  bf16_t* w_ff1 = wr[10].dst; bf16_t* w_ff2 = wr[11].dst;
  bf16_t* w_ffemb = wr[12].dst; bf16_t* w_ffout = wr[13].dst;

  bf16_t* actbf = (bf16_t*)carve(BT * Dc * 2);      // LN / FiLM / x-bf16 staging
  bf16_t* tnbf  = (bf16_t*)carve(BNr * Dtc * 2);
  float*  qfb   = (float*)carve(BT * Dc * 4);
  float*  kfb   = (float*)carve(BT * Dc * 4);
  float*  vfb   = (float*)carve(BT * Dc * 4);
  bf16_t* qbf   = (bf16_t*)carve(BT * Dc * 2);
  bf16_t* kTb   = (bf16_t*)carve(BT * Dc * 2);      // [B,H,64,T] (SA) / [B,H,64,N] (CA)
  bf16_t* vbf   = (bf16_t*)carve(BT * Dc * 2);
  bf16_t* attb  = (bf16_t*)carve((long)Bc * Hc * 64 * 64 * 2);
  float*  yfb   = (float*)carve(BT * Dc * 4);
  float*  eo    = (float*)carve((long)Bc * 2 * Dc * 4);
  bf16_t* embs  = (bf16_t*)carve((long)Bc * TEc * 2);
  float*  x1    = (float*)carve(BT * Dc * 4);
  float*  x2    = (float*)carve(BT * Dc * 4);
  bf16_t* ffm   = (bf16_t*)carve(BT * FFc * 2);

  // ---- weight conversion f32 -> bf16 ----
  for (int i = 0; i < 14; i++)
    cvt_bf16_k<<<dim3(1024), dim3(256), 0, stream>>>(P[wr[i].idx], wr[i].dst, wr[i].n);

  // shared emb SiLU (used by all three styl blocks)
  silu_bf16_k<<<dim3(64), dim3(256), 0, stream>>>(emb, embs, (long)Bc * TEc);

  // ===================== self-attention block =====================
  ln_film_k<<<dim3((unsigned)BT), dim3(256), 0, stream>>>(x, P[0], P[1], nullptr, 1, actbf, Dc, 0);
  run_gemm(stream, 0, mk2d(actbf, Dc, w_saq, Dc, qfb, Dc, P[3], nullptr, 0, (int)BT, Dc, Dc), 1, 128);
  run_gemm(stream, 0, mk2d(actbf, Dc, w_sak, Dc, kfb, Dc, P[5], nullptr, 0, (int)BT, Dc, Dc), 1, 128);
  run_gemm(stream, 0, mk2d(actbf, Dc, w_sav, Dc, vfb, Dc, P[7], nullptr, 0, (int)BT, Dc, Dc), 1, 128);
  softmax_hd_k<<<dim3(32768), dim3(256), 0, stream>>>(qfb, qbf, BT * Hc);
  softmax_t_tr_k<<<dim3(Bc * Hc), dim3(64), 0, stream>>>(kfb, mask, kTb, Tc);
  maskmul_bf16_k<<<dim3(4096), dim3(256), 0, stream>>>(vfb, mask, vbf, BT * Dc, Dc);
  {  // att[bh] = kT(64 x T) @ v(T x 64)
    GemmArgs g = mk2d(kTb, Tc, vbf, Dc, attb, 64, nullptr, nullptr, 0, 64, 64, Tc);
    g.sA1 = (long)64 * Tc;
    g.sB1 = (long)Tc * Dc; g.sB2 = 64; g.nbB2 = Hc;
    g.sC1 = 64 * 64;
    run_gemm(stream, 1, g, Bc * Hc, 64);
  }
  {  // y[bh] = q(T x 64) @ att(64 x 64)
    GemmArgs g = mk2d(qbf, Dc, attb, 64, yfb, Dc, nullptr, nullptr, 0, Tc, 64, 64);
    g.sA1 = (long)Tc * Dc; g.sA2 = 64; g.nbA2 = Hc;
    g.sB1 = 64 * 64;
    g.sC1 = (long)Tc * Dc; g.sC2 = 64; g.nbC2 = Hc;
    run_gemm(stream, 0, g, Bc * Hc, 64);
  }
  run_gemm(stream, 0, mk2d(embs, TEc, w_saemb, 2 * Dc, eo, 2 * Dc, P[9], nullptr, 0, Bc, 2 * Dc, TEc), 1, 128);
  ln_film_k<<<dim3((unsigned)BT), dim3(256), 0, stream>>>(yfb, P[10], P[11], eo, Tc, actbf, Dc, 1);
  run_gemm(stream, 3, mk2d(actbf, Dc, w_saout, Dc, x1, Dc, P[13], x, Dc, (int)BT, Dc, Dc), 1, 128);

  // ===================== cross-attention block =====================
  ln_film_k<<<dim3((unsigned)BT), dim3(256), 0, stream>>>(x1, P[14], P[15], nullptr, 1, actbf, Dc, 0);
  ln_film_k<<<dim3((unsigned)BNr), dim3(256), 0, stream>>>(xf, P[16], P[17], nullptr, 1, tnbf, Dtc, 0);
  run_gemm(stream, 0, mk2d(actbf, Dc, w_caq, Dc, qfb, Dc, P[19], nullptr, 0, (int)BT, Dc, Dc), 1, 128);
  run_gemm(stream, 0, mk2d(tnbf, Dtc, w_cak, Dc, kfb, Dc, P[21], nullptr, 0, (int)BNr, Dc, Dtc), 1, 128);
  run_gemm(stream, 0, mk2d(tnbf, Dtc, w_cav, Dc, vfb, Dc, P[23], nullptr, 0, (int)BNr, Dc, Dtc), 1, 128);
  softmax_hd_k<<<dim3(32768), dim3(256), 0, stream>>>(qfb, qbf, BT * Hc);
  softmax_t_tr_k<<<dim3(Bc * Hc), dim3(64), 0, stream>>>(kfb, nullptr, kTb, Nc);
  maskmul_bf16_k<<<dim3(4096), dim3(256), 0, stream>>>(vfb, nullptr, vbf, BNr * Dc, Dc);
  {
    GemmArgs g = mk2d(kTb, Nc, vbf, Dc, attb, 64, nullptr, nullptr, 0, 64, 64, Nc);
    g.sA1 = (long)64 * Nc;
    g.sB1 = (long)Nc * Dc; g.sB2 = 64; g.nbB2 = Hc;
    g.sC1 = 64 * 64;
    run_gemm(stream, 1, g, Bc * Hc, 64);
  }
  {
    GemmArgs g = mk2d(qbf, Dc, attb, 64, yfb, Dc, nullptr, nullptr, 0, Tc, 64, 64);
    g.sA1 = (long)Tc * Dc; g.sA2 = 64; g.nbA2 = Hc;
    g.sB1 = 64 * 64;
    g.sC1 = (long)Tc * Dc; g.sC2 = 64; g.nbC2 = Hc;
    run_gemm(stream, 0, g, Bc * Hc, 64);
  }
  run_gemm(stream, 0, mk2d(embs, TEc, w_caemb, 2 * Dc, eo, 2 * Dc, P[25], nullptr, 0, Bc, 2 * Dc, TEc), 1, 128);
  ln_film_k<<<dim3((unsigned)BT), dim3(256), 0, stream>>>(yfb, P[26], P[27], eo, Tc, actbf, Dc, 1);
  run_gemm(stream, 3, mk2d(actbf, Dc, w_caout, Dc, x2, Dc, P[29], x1, Dc, (int)BT, Dc, Dc), 1, 128);

  // ===================== FFN block =====================
  cvt_bf16_k<<<dim3(4096), dim3(256), 0, stream>>>(x2, actbf, BT * Dc);
  run_gemm(stream, 2, mk2d(actbf, Dc, w_ff1, FFc, ffm, FFc, P[31], nullptr, 0, (int)BT, FFc, Dc), 1, 128);
  run_gemm(stream, 0, mk2d(ffm, FFc, w_ff2, Dc, yfb, Dc, P[33], nullptr, 0, (int)BT, Dc, FFc), 1, 128);
  run_gemm(stream, 0, mk2d(embs, TEc, w_ffemb, 2 * Dc, eo, 2 * Dc, P[35], nullptr, 0, Bc, 2 * Dc, TEc), 1, 128);
  ln_film_k<<<dim3((unsigned)BT), dim3(256), 0, stream>>>(yfb, P[36], P[37], eo, Tc, actbf, Dc, 1);
  run_gemm(stream, 3, mk2d(actbf, Dc, w_ffout, Dc, (float*)d_out, Dc, P[39], x2, Dc, (int)BT, Dc, Dc), 1, 128);
}
